// GRUDClassifier_67877663146571
// MI455X (gfx1250) — matrix operand order repaired and checked by reading, not compiled, not run on hardware
//
#include <hip/hip_runtime.h>
#include <hip/hip_bf16.h>

// ---------------------------------------------------------------------------
// GRU-D  (B=64, I=128, H=256, T=1024) for gfx1250 (CDNA5, wave32, WMMA)
//
// Phase 0: pack weights into WMMA bf16 B-fragment layout (one 32B load/lane)
// Phase 1: tiled transpose [B,I,T]->[T,B,I]; gamma GEMMs (bf16 WMMA, fused
//          exp(-relu(.+b))); x_last scan + x2; one fused GEMM
//          [T*B,256]x[256,768] -> Ax = {z|r|h} pre-activations (+bias)
// Phase 2: persistent single-workgroup recurrence: Whz in LDS, Whr/Whh
//          register-resident per wave, bf16 h/hd/r*hd staging in LDS,
//          96 v_wmma per step with split accumulator chains (no hazard NOPs).
// ---------------------------------------------------------------------------

#define B_ 64
#define I_ 128
#define H_ 256
#define T_ 1024

typedef __attribute__((ext_vector_type(16))) __bf16 v16bf;
typedef __attribute__((ext_vector_type(8)))  float  v8f;
typedef __attribute__((ext_vector_type(4)))  unsigned int u32x4;

union FragAB { v16bf v; u32x4 q[2]; };

__device__ __forceinline__ v8f wmma_bf16(v16bf a, v16bf b, v8f c) {
  return __builtin_amdgcn_wmma_f32_16x16x32_bf16(false, a, false, b,
                                                 (short)0, c, false, false);
}

__device__ __forceinline__ v16bf load_frag_bytes(const char* p) {
  FragAB f;
  f.q[0] = *(const u32x4*)(p);
  f.q[1] = *(const u32x4*)(p + 16);
  return f.v;
}

// ---------------------------------------------------------------------------
// Weight packing: B[k][n] = W[n][k] into WMMA B-fragment lane order.
// Element j of lane L in frag (nb,kb):  K = kb*32 + (L/16)*16 + j,
//                                       N = nb*16 + L%16.
// ---------------------------------------------------------------------------
__global__ void pack_bt_kernel(const float* __restrict__ W,
                               __hip_bfloat16* __restrict__ dst, int K, int N) {
  int gid = blockIdx.x * blockDim.x + threadIdx.x;
  int Kb = K >> 5, Nb = N >> 4;
  if (gid >= Nb * Kb * 32) return;
  int lane = gid & 31, frag = gid >> 5;
  int nb = frag / Kb, kb = frag % Kb;
  int n  = nb * 16 + (lane & 15);
  int k0 = kb * 32 + ((lane & 16) ? 16 : 0);
#pragma unroll
  for (int j = 0; j < 16; ++j)
    dst[(size_t)gid * 16 + j] = __float2bfloat16(W[(size_t)n * K + k0 + j]);
}

// Concatenated-K pack for [Wx* | Wm*]: K=256 (0..127 from W1, 128..255 from W2),
// gate slice placed at nbBase within the combined N=768 output.
__global__ void pack_bt2_kernel(const float* __restrict__ W1,
                                const float* __restrict__ W2,
                                __hip_bfloat16* __restrict__ dst, int nbBase) {
  int gid = blockIdx.x * blockDim.x + threadIdx.x;
  if (gid >= 16 * 8 * 32) return;
  int lane = gid & 31, frag = gid >> 5;
  int nb = frag >> 3, kb = frag & 7;
  int n  = nb * 16 + (lane & 15);
  int k0 = kb * 32 + ((lane & 16) ? 16 : 0);
  size_t dbase = ((size_t)((nbBase + nb) * 8 + kb) * 32 + lane) * 16;
#pragma unroll
  for (int j = 0; j < 16; ++j) {
    int k = k0 + j;
    float v = (k < 128) ? W1[(size_t)n * 128 + k] : W2[(size_t)n * 128 + (k - 128)];
    dst[dbase + j] = __float2bfloat16(v);
  }
}

__global__ void build_bias_kernel(const float* __restrict__ bmz,
                                  const float* __restrict__ bmh,
                                  float* __restrict__ biasAx) {
  int gid = blockIdx.x * blockDim.x + threadIdx.x;
  if (gid < 768)
    biasAx[gid] = (gid < 256) ? bmz[gid] : ((gid < 512) ? 0.f : bmh[gid - 512]);
}

// ---------------------------------------------------------------------------
// LDS-tiled transpose [B,I,T] -> [T,B,I]; also fuses mask into the GEMM
// A-matrix (bf16) and converts Delta to bf16.
// ---------------------------------------------------------------------------
__global__ void transpose_in_kernel(const float* __restrict__ X,
                                    const float* __restrict__ Mask,
                                    const float* __restrict__ Delta,
                                    float* __restrict__ Xt,
                                    float* __restrict__ Mt,
                                    __hip_bfloat16* __restrict__ Dbf,
                                    __hip_bfloat16* __restrict__ Axm) {
  __shared__ float tile[I_][33];
  int b = blockIdx.y, t0 = blockIdx.x * 32;
  int tid = threadIdx.x, lt = tid & 31, i0 = tid >> 5;
  for (int pass = 0; pass < 3; ++pass) {
    const float* src = (pass == 0) ? X : (pass == 1) ? Mask : Delta;
    for (int i = i0; i < I_; i += 8)
      tile[i][lt] = src[((size_t)b * I_ + i) * T_ + t0 + lt];
    __syncthreads();
    for (int e = tid; e < 32 * I_; e += 256) {
      int tt = e >> 7, ii = e & 127;
      float v = tile[ii][tt];
      size_t r = (size_t)(t0 + tt) * B_ + b;
      if (pass == 0)      Xt[r * I_ + ii] = v;
      else if (pass == 1) { Mt[r * I_ + ii] = v;
                            Axm[r * 256 + 128 + ii] = __float2bfloat16(v); }
      else                Dbf[r * I_ + ii] = __float2bfloat16(v);
    }
    __syncthreads();
  }
}

// ---------------------------------------------------------------------------
// WMMA GEMM: D[M, Nb*16] = A[M, Kb*32] x Bpacked (+bias).
// Each wave computes a 16M x 64N strip (4 N-tiles): one A-frag load feeds 4
// independent WMMA chains -> no WMMA->WMMA hazard NOPs, 4x A reuse.
// mode 0: f32 out (acc + bias). mode 1: bf16 out = exp(-relu(acc + bias)).
// ---------------------------------------------------------------------------
__global__ void gemm_bf16_kernel(const __hip_bfloat16* __restrict__ A,
                                 const __hip_bfloat16* __restrict__ Bp,
                                 const float* __restrict__ bias,
                                 float* __restrict__ Df32,
                                 __hip_bfloat16* __restrict__ Dbf,
                                 int Mtiles, int Kb, int Nb, int mode) {
  int wid  = (blockIdx.x * blockDim.x + threadIdx.x) >> 5;
  int lane = threadIdx.x & 31;
  int Nq = Nb >> 2;                      // groups of 4 N-tiles
  if (wid >= Mtiles * Nq) return;
  int mt = wid / Nq, nb0 = (wid % Nq) * 4;
  int ldA = Kb * 32, ldD = Nb * 16;
  int row = mt * 16 + (lane & 15);

  // Per-wave base pointers; inner loop uses constant byte offsets.
  const char* aptr = (const char*)(A + (size_t)row * ldA + ((lane & 16) ? 16 : 0));
  const size_t bstride = (size_t)Kb * 512;               // bytes per N-tile stream
  const char* bptr = (const char*)Bp + (size_t)nb0 * bstride + (size_t)lane * 32;

  v8f acc0 = {}, acc1 = {}, acc2 = {}, acc3 = {};
  for (int kb = 0; kb < Kb; ++kb) {
    v16bf a  = load_frag_bytes(aptr + (size_t)kb * 64);
    size_t bo = (size_t)kb * 512;
    v16bf b0 = load_frag_bytes(bptr + bo);
    v16bf b1 = load_frag_bytes(bptr + bstride + bo);
    v16bf b2 = load_frag_bytes(bptr + 2 * bstride + bo);
    v16bf b3 = load_frag_bytes(bptr + 3 * bstride + bo);
    acc0 = wmma_bf16(a, b0, acc0);
    acc1 = wmma_bf16(a, b1, acc1);
    acc2 = wmma_bf16(a, b2, acc2);
    acc3 = wmma_bf16(a, b3, acc3);
  }

  int mhalf = (lane & 16) ? 8 : 0;
  v8f accs[4] = {acc0, acc1, acc2, acc3};
#pragma unroll
  for (int q = 0; q < 4; ++q) {
    int n = (nb0 + q) * 16 + (lane & 15);
    float bval = bias ? bias[n] : 0.f;
#pragma unroll
    for (int j = 0; j < 8; ++j) {
      int m = mt * 16 + mhalf + j;
      float v = accs[q][j] + bval;
      if (mode == 0) Df32[(size_t)m * ldD + n] = v;
      else           Dbf[(size_t)m * ldD + n] =
                         __float2bfloat16(__expf(-fmaxf(v, 0.f)));
    }
  }
}

// ---------------------------------------------------------------------------
// x_last scan + x2 imputation (binary mask): x2 = m ? x : gx*xl + (1-gx)*xm.
// Operates in transposed layout -> fully coalesced over i.
// ---------------------------------------------------------------------------
__global__ void scan_x2_kernel(const float* __restrict__ Xt,
                               const float* __restrict__ Mt,
                               const __hip_bfloat16* __restrict__ Gx,
                               const float* __restrict__ xmean_p,
                               __hip_bfloat16* __restrict__ Axm) {
  int gid = blockIdx.x * blockDim.x + threadIdx.x;  // B_*I_ threads
  int b = gid >> 7, i = gid & 127;
  float xm = *xmean_p;
  float xl = 0.f;
  for (int t = 0; t < T_; ++t) {
    size_t base = ((size_t)t * B_ + b) * I_ + i;
    float m = Mt[base], x = Xt[base];
    float x2;
    if (m > 0.f) { xl = x; x2 = x; }
    else { float g = __bfloat162float(Gx[base]); x2 = g * xl + (1.f - g) * xm; }
    Axm[((size_t)t * B_ + b) * 256 + i] = __float2bfloat16(x2);
  }
}

// ---------------------------------------------------------------------------
// Phase 2: persistent recurrence. 1 workgroup, 512 threads = 16 waves.
// Wave w owns output columns [16w, 16w+16). Whz in LDS (frag order),
// Whr/Whh register-resident (8 v16bf each). Dynamic LDS = 224 KB.
// Accumulator chains are split (z0/z1/r0/r1 and h0/h1) so dependent WMMAs
// are >= 4 issue slots apart -> the hazard window is filled with real work.
// ---------------------------------------------------------------------------
__global__ void __launch_bounds__(512)
grud_recurrent_kernel(const __hip_bfloat16* __restrict__ Gh,   // [T][B][H] bf16
                      const float* __restrict__ Ax,            // [T*B][768] f32
                      const __hip_bfloat16* __restrict__ BpHz,
                      const __hip_bfloat16* __restrict__ BpHr,
                      const __hip_bfloat16* __restrict__ BpHh,
                      float* __restrict__ Out) {               // [B][T][H]
  extern __shared__ char smem[];
  __hip_bfloat16* ldsWhz = (__hip_bfloat16*)smem;              // 64K elems (128KB)
  __hip_bfloat16* Hbuf   = ldsWhz + H_ * H_;                   // 16K elems (32KB)
  __hip_bfloat16* HDbuf  = Hbuf + B_ * H_;
  __hip_bfloat16* RHDbuf = HDbuf + B_ * H_;

  int tid = threadIdx.x, lane = tid & 31, w = tid >> 5;

  // Stage Whz into LDS (frag order mirrors global packing).
  {
    const u32x4* src = (const u32x4*)BpHz;
    u32x4* dst = (u32x4*)ldsWhz;
    for (int o = tid; o < (H_ * H_ * 2) / 16; o += 512) dst[o] = src[o];
  }
  // h0 = 0
  for (int o = tid; o < (B_ * H_) / 2; o += 512) ((unsigned int*)Hbuf)[o] = 0u;

  // Register-resident Whr / Whh fragments for this wave's N-slice.
  v16bf whr[8], whh[8];
#pragma unroll
  for (int kb = 0; kb < 8; ++kb) {
    whr[kb] = load_frag_bytes((const char*)BpHr +
                              ((size_t)(w * 8 + kb) * 32 + lane) * 32);
    whh[kb] = load_frag_bytes((const char*)BpHh +
                              ((size_t)(w * 8 + kb) * 32 + lane) * 32);
  }
  __syncthreads();

  const int nlane = w * 16 + (lane & 15);
  const int mhalf = (lane & 16) ? 8 : 0;
  const char* whzLds = (const char*)ldsWhz + ((size_t)w * 8 * 32 + lane) * 32;
  v8f zsave[4];

  for (int t = 0; t < T_; ++t) {
    const __hip_bfloat16* ghRow = Gh + (size_t)t * B_ * H_;
    const float* axRow = Ax + (size_t)t * B_ * 768;
    if (t + 1 < T_) {
      __builtin_prefetch(Gh + (size_t)(t + 1) * B_ * H_ + tid * 32, 0, 1);
      __builtin_prefetch(Ax + (size_t)(t + 1) * B_ * 768 + tid * 96, 0, 1);
    }

    // Stage A: hd = gamma_h * h  (bf16 into LDS)
    for (int e = tid; e < B_ * H_; e += 512) {
      float hv = __bfloat162float(Hbuf[e]);
      float g  = __bfloat162float(ghRow[e]);
      HDbuf[e] = __float2bfloat16(g * hv);
    }
    __syncthreads();

    // Stage B: z = sigmoid(Axz + hd@Whz^T), r = sigmoid(Axr + hd@Whr^T)
#pragma unroll
    for (int mt = 0; mt < 4; ++mt) {
      v8f accz0 = {}, accz1 = {}, accr0 = {}, accr1 = {};
      const char* arowp = (const char*)(HDbuf + (size_t)(mt * 16 + (lane & 15)) * H_
                                        + ((lane & 16) ? 16 : 0));
#pragma unroll
      for (int kb = 0; kb < 8; kb += 2) {
        v16bf a0 = load_frag_bytes(arowp + (size_t)kb * 64);
        v16bf a1 = load_frag_bytes(arowp + (size_t)(kb + 1) * 64);
        v16bf bz0 = load_frag_bytes(whzLds + (size_t)kb * 1024);
        v16bf bz1 = load_frag_bytes(whzLds + (size_t)(kb + 1) * 1024);
        accz0 = wmma_bf16(a0, bz0, accz0);
        accr0 = wmma_bf16(a0, whr[kb], accr0);
        accz1 = wmma_bf16(a1, bz1, accz1);
        accr1 = wmma_bf16(a1, whr[kb + 1], accr1);
      }
      v8f accz = accz0 + accz1;
      v8f accr = accr0 + accr1;
#pragma unroll
      for (int j = 0; j < 8; ++j) {
        int b = mt * 16 + mhalf + j;
        size_t axb = (size_t)b * 768 + nlane;
        float z = 1.f / (1.f + __expf(-(axRow[axb] + accz[j])));
        float r = 1.f / (1.f + __expf(-(axRow[axb + 256] + accr[j])));
        float hd = __bfloat162float(HDbuf[b * H_ + nlane]);
        RHDbuf[b * H_ + nlane] = __float2bfloat16(r * hd);
        zsave[mt][j] = z;
      }
    }
    __syncthreads();

    // Stage C: h~ = tanh(Axh + (r*hd)@Whh^T); h = (1-z)*hd + z*h~
#pragma unroll
    for (int mt = 0; mt < 4; ++mt) {
      v8f acch0 = {}, acch1 = {};
      const char* arowp = (const char*)(RHDbuf + (size_t)(mt * 16 + (lane & 15)) * H_
                                        + ((lane & 16) ? 16 : 0));
#pragma unroll
      for (int kb = 0; kb < 8; kb += 2) {
        v16bf a0 = load_frag_bytes(arowp + (size_t)kb * 64);
        v16bf a1 = load_frag_bytes(arowp + (size_t)(kb + 1) * 64);
        acch0 = wmma_bf16(a0, whh[kb], acch0);
        acch1 = wmma_bf16(a1, whh[kb + 1], acch1);
      }
      v8f acch = acch0 + acch1;
#pragma unroll
      for (int j = 0; j < 8; ++j) {
        int b = mt * 16 + mhalf + j;
        float ht = tanhf(axRow[(size_t)b * 768 + 512 + nlane] + acch[j]);
        float hd = __bfloat162float(HDbuf[b * H_ + nlane]);
        float z = zsave[mt][j];
        float hn = (1.f - z) * hd + z * ht;
        Out[((size_t)b * T_ + t) * H_ + nlane] = hn;
        Hbuf[b * H_ + nlane] = __float2bfloat16(hn);
      }
    }
    __syncthreads();
  }
}

// ---------------------------------------------------------------------------
extern "C" void kernel_launch(void* const* d_in, const int* in_sizes, int n_in,
                              void* d_out, int out_size, void* d_ws, size_t ws_size,
                              hipStream_t stream) {
  const float* X     = (const float*)d_in[0];
  const float* Mask  = (const float*)d_in[1];
  const float* Delta = (const float*)d_in[2];
  const float* xmean = (const float*)d_in[3];
  const float* Wdgx  = (const float*)d_in[4];
  const float* bdgx  = (const float*)d_in[5];
  const float* Wdgh  = (const float*)d_in[6];
  const float* bdgh  = (const float*)d_in[7];
  const float* Wxz   = (const float*)d_in[8];
  const float* Whz   = (const float*)d_in[9];
  const float* Wmz   = (const float*)d_in[10];
  const float* bmz   = (const float*)d_in[11];
  const float* Wxr   = (const float*)d_in[12];
  const float* Whr   = (const float*)d_in[13];
  const float* Wmr   = (const float*)d_in[14];
  const float* Wxh   = (const float*)d_in[15];
  const float* Whh   = (const float*)d_in[16];
  const float* Wmh   = (const float*)d_in[17];
  const float* bmh   = (const float*)d_in[18];
  float* Out = (float*)d_out;

  const size_t TB = (size_t)T_ * B_;
  char* ws = (char*)d_ws;
  size_t off = 0;
  auto alloc = [&](size_t bytes) -> void* {
    void* p = ws + off;
    off = (off + bytes + 255) & ~(size_t)255;
    return p;
  };
  float*          Xt     = (float*)alloc(TB * I_ * 4);
  float*          Mt     = (float*)alloc(TB * I_ * 4);
  __hip_bfloat16* Dbf    = (__hip_bfloat16*)alloc(TB * I_ * 2);
  __hip_bfloat16* Gx     = (__hip_bfloat16*)alloc(TB * I_ * 2);
  __hip_bfloat16* Gh     = (__hip_bfloat16*)alloc(TB * H_ * 2);
  __hip_bfloat16* Axm    = (__hip_bfloat16*)alloc(TB * 256 * 2);
  float*          Ax     = (float*)alloc(TB * 768 * 4);
  __hip_bfloat16* BpDgx  = (__hip_bfloat16*)alloc(128 * 128 * 2);
  __hip_bfloat16* BpDgh  = (__hip_bfloat16*)alloc(128 * 256 * 2);
  __hip_bfloat16* BpXm   = (__hip_bfloat16*)alloc(256 * 768 * 2);
  __hip_bfloat16* BpHz   = (__hip_bfloat16*)alloc(256 * 256 * 2);
  __hip_bfloat16* BpHr   = (__hip_bfloat16*)alloc(256 * 256 * 2);
  __hip_bfloat16* BpHh   = (__hip_bfloat16*)alloc(256 * 256 * 2);
  float*          biasAx = (float*)alloc(768 * 4);

  // Phase 0: weight packing + bias
  pack_bt_kernel<<<4,  256, 0, stream>>>(Wdgx, BpDgx, 128, 128);
  pack_bt_kernel<<<8,  256, 0, stream>>>(Wdgh, BpDgh, 128, 256);
  pack_bt_kernel<<<16, 256, 0, stream>>>(Whz,  BpHz,  256, 256);
  pack_bt_kernel<<<16, 256, 0, stream>>>(Whr,  BpHr,  256, 256);
  pack_bt_kernel<<<16, 256, 0, stream>>>(Whh,  BpHh,  256, 256);
  pack_bt2_kernel<<<16, 256, 0, stream>>>(Wxz, Wmz, BpXm, 0);
  pack_bt2_kernel<<<16, 256, 0, stream>>>(Wxr, Wmr, BpXm, 16);
  pack_bt2_kernel<<<16, 256, 0, stream>>>(Wxh, Wmh, BpXm, 32);
  build_bias_kernel<<<3, 256, 0, stream>>>(bmz, bmh, biasAx);

  // Phase 1: transpose inputs
  transpose_in_kernel<<<dim3(T_ / 32, B_), 256, 0, stream>>>(
      X, Mask, Delta, Xt, Mt, Dbf, Axm);

  // gamma_x = exp(-relu(D @ Wdgx^T + bdgx))   [65536 x 128], 4096*2 waves
  gemm_bf16_kernel<<<1024, 256, 0, stream>>>(Dbf, BpDgx, bdgx, nullptr, Gx,
                                             4096, 4, 8, 1);
  // gamma_h = exp(-relu(D @ Wdgh^T + bdgh))   [65536 x 256], 4096*4 waves
  gemm_bf16_kernel<<<2048, 256, 0, stream>>>(Dbf, BpDgh, bdgh, nullptr, Gh,
                                             4096, 4, 16, 1);
  // x_last scan + x2 imputation -> Axm[:, 0:128]
  scan_x2_kernel<<<32, 256, 0, stream>>>(Xt, Mt, Gx, xmean, Axm);
  // Ax = [x2|m] @ [Wxz|Wmz ; Wxr|Wmr ; Wxh|Wmh]^T + [bmz|0|bmh]  [65536 x 768]
  gemm_bf16_kernel<<<6144, 256, 0, stream>>>(Axm, BpXm, biasAx, Ax, nullptr,
                                             4096, 8, 48, 0);

  // Phase 2: persistent recurrence (224 KB dynamic LDS)
  size_t shbytes = (size_t)(H_ * H_ + 3 * B_ * H_) * 2;
  grud_recurrent_kernel<<<1, 512, shbytes, stream>>>(Gh, Ax, BpHz, BpHr, BpHh, Out);
}
